// DGCNN_9388798509748
// MI455X (gfx1250) — matrix operand order repaired and checked by reading, compiled, not run on hardware
//
#include <hip/hip_runtime.h>

#define KNN 20
#define BB 16
#define NN 2048

typedef __attribute__((ext_vector_type(16))) _Float16 v16h;
typedef __attribute__((ext_vector_type(8)))  _Float16 v8h;
typedef __attribute__((ext_vector_type(8)))  float    v8f;

// ---------------- WMMA fragment helpers (16x16x32 f16; contiguous k order, same for A and B) ----------------
// A 16x32 (MxK): lanes 0-15 -> M=lane, K=0..15 contiguous; lanes 16-31 -> K=16..31
__device__ __forceinline__ v16h load_a_frag(const _Float16* tile, int ld, int lane) {
    int m = lane & 15;
    int k = (lane >> 4) << 4;
    const _Float16* row = tile + m * ld;
    v8h lo = *(const v8h*)(row + k);
    v8h hi = *(const v8h*)(row + k + 8);
    v16h r;
#pragma unroll
    for (int i = 0; i < 8; ++i) { r[i] = lo[i]; r[i + 8] = hi[i]; }
    return r;
}
// B 32x16 (KxN): lanes 0-15 -> N=lane, K=0..15 contiguous; lanes 16-31 -> K=16..31
__device__ __forceinline__ v16h load_b_frag(const _Float16* tile, int ld, int lane) {
    int n = lane & 15;
    int k = (lane >> 4) << 4;
    return *(const v16h*)(tile + n * ld + k);
}
__device__ __forceinline__ v8f wmma16(v16h a, v16h b, v8f c) {
    return __builtin_amdgcn_wmma_f32_16x16x32_f16(false, a, false, b, (short)0, c, false, false);
}

// ---------------- CDNA5 async global->LDS copy (ASYNCcnt path) ----------------
// LDS byte address = low 32 bits of the flat pointer (LDS aperture mapping, ISA 10.2).
__device__ __forceinline__ void async_cp16(void* lds_dst, const void* gsrc) {
    unsigned int l = (unsigned int)(unsigned long long)lds_dst;
    asm volatile("global_load_async_to_lds_b128 %0, %1, off"
                 :: "v"(l), "v"(gsrc) : "memory");
}
__device__ __forceinline__ void async_wait0() {
    asm volatile("s_wait_asynccnt 0x0" ::: "memory");
}
__device__ __forceinline__ void async_wait2() {
    asm volatile("s_wait_asynccnt 0x2" ::: "memory");
}
__device__ __forceinline__ void async_wait6() {
    asm volatile("s_wait_asynccnt 0x6" ::: "memory");
}

// ---------------- small utility kernels ----------------
__global__ void fill_kernel(float* p, int n, float v) {
    int i = blockIdx.x * blockDim.x + threadIdx.x;
    if (i < n) p[i] = v;
}

// x (B,3,N) fp32 -> xh0 (B,N,32) f16, zero padded
__global__ void conv_in_kernel(const float* __restrict__ x, _Float16* __restrict__ xh0) {
    int i = blockIdx.x * blockDim.x + threadIdx.x;    // over B*N
    if (i >= BB * NN) return;
    int b = i / NN, n = i - b * NN;
#pragma unroll
    for (int c = 0; c < 32; ++c)
        xh0[(size_t)i * 32 + c] = (c < 3) ? (_Float16)x[((size_t)b * 3 + c) * NN + n] : (_Float16)0.f;
}

// Split W (O x 2C) into f16 Wd = W[:, :C] and Wdelta = W[:, C:] - W[:, :C], padded to Cpad
__global__ void prep_w_kernel(const float* __restrict__ W, _Float16* __restrict__ wd,
                              _Float16* __restrict__ wdel, int O, int C, int Cpad) {
    int i = blockIdx.x * blockDim.x + threadIdx.x;    // over O*Cpad
    if (i >= O * Cpad) return;
    int o = i / Cpad, c = i - o * Cpad;
    float a = 0.f, d = 0.f;
    if (c < C) {
        a = W[(size_t)o * 2 * C + c];
        d = W[(size_t)o * 2 * C + C + c] - a;
    }
    wd[i] = (_Float16)a;
    wdel[i] = (_Float16)d;
}

__global__ void conv_w5_kernel(const float* __restrict__ W5, _Float16* __restrict__ w5h) {
    int i = blockIdx.x * blockDim.x + threadIdx.x;    // over 1024*512
    if (i < 1024 * 512) w5h[i] = (_Float16)W5[i];
}

__global__ void sqnorm_kernel(const _Float16* __restrict__ feat, int stride, int chanoff, int Cpad,
                              float* __restrict__ sq) {
    int i = blockIdx.x * blockDim.x + threadIdx.x;    // over B*N
    if (i >= BB * NN) return;
    const _Float16* row = feat + (size_t)i * stride + chanoff;
    float s = 0.f;
    for (int c = 0; c < Cpad; ++c) { float v = (float)row[c]; s += v * v; }
    sq[i] = s;
}

__global__ void bn_finalize_kernel(const float* __restrict__ sum, const float* __restrict__ sumsq,
                                   const float* __restrict__ gamma, const float* __restrict__ beta,
                                   float* __restrict__ scale, float* __restrict__ shift,
                                   int O, float invcnt) {
    int o = blockIdx.x * blockDim.x + threadIdx.x;
    if (o >= O) return;
    float mean = sum[o] * invcnt;
    float var  = sumsq[o] * invcnt - mean * mean;
    float s = gamma[o] * rsqrtf(var + 1e-5f);
    scale[o] = s;
    shift[o] = beta[o] - mean * s;
}

// ---------------- pairwise -dist^2 via WMMA, double-buffered async staging ----------------
__global__ __launch_bounds__(256)
void dist_kernel(const _Float16* __restrict__ feat, int stride, int chanoff, int Cpad,
                 const float* __restrict__ sq, float* __restrict__ dist) {
    __shared__ __align__(64) _Float16 xi[2 * 64 * 32];
    __shared__ __align__(64) _Float16 xj[2 * 64 * 32];
    __shared__ float sqi[64], sqj[64];
    int b = blockIdx.z;
    int ibase = blockIdx.x * 64;
    int jbase = blockIdx.y * 64;
    int tid = threadIdx.x, lane = tid & 31, wave = tid >> 5;

    if (tid < 64) sqi[tid] = sq[b * NN + ibase + tid];
    else if (tid < 128) sqj[tid - 64] = sq[b * NN + jbase + (tid - 64)];

    v8f acc0 = {}, acc1 = {};
    const int t0 = wave, t1 = wave + 8;          // 16 tiles (4x4) / 8 waves
    const _Float16* fb = feat + (size_t)b * NN * stride + chanoff;
    int nchunks = Cpad >> 5;
    int r = tid >> 2, q = (tid & 3) * 8;         // 64 rows x 4 quads of 8 halves

    auto issue = [&](int kc, int p) {            // 2 async copies / thread / chunk
        int c0 = kc << 5;
        async_cp16(&xi[p * 2048 + r * 32 + q], fb + (size_t)(ibase + r) * stride + c0 + q);
        async_cp16(&xj[p * 2048 + r * 32 + q], fb + (size_t)(jbase + r) * stride + c0 + q);
    };
    issue(0, 0);
    for (int kc = 0; kc < nchunks; ++kc) {
        int p = kc & 1;
        __syncthreads();                          // buf[(kc+1)&1] reads (iter kc-1) retired
        if (kc + 1 < nchunks) { issue(kc + 1, p ^ 1); async_wait2(); }
        else                  { async_wait0(); }
        __syncthreads();                          // chunk kc landed block-wide
        const _Float16* bi = xi + p * 2048;
        const _Float16* bj = xj + p * 2048;
        {
            int ti = t0 >> 2, tj = t0 & 3;
            acc0 = wmma16(load_a_frag(bi + ti * 16 * 32, 32, lane),
                          load_b_frag(bj + tj * 16 * 32, 32, lane), acc0);
        }
        {
            int ti = t1 >> 2, tj = t1 & 3;
            acc1 = wmma16(load_a_frag(bi + ti * 16 * 32, 32, lane),
                          load_b_frag(bj + tj * 16 * 32, 32, lane), acc1);
        }
    }
    int hi = (lane >> 4) * 8, col = lane & 15;
#pragma unroll
    for (int s = 0; s < 2; ++s) {
        v8f acc = s ? acc1 : acc0;
        int t = s ? t1 : t0;
        int ti = t >> 2, tj = t & 3;
#pragma unroll
        for (int rr = 0; rr < 8; ++rr) {
            int li = ti * 16 + rr + hi;
            int lj = tj * 16 + col;
            int gi = ibase + li, gj = jbase + lj;
            float v = 2.f * acc[rr] - sqi[li] - sqj[lj];
            if (gi == gj) v = 0.f;
            // streaming 268MB matrix: NT store keeps L2 (192MB) for resident data
            __builtin_nontemporal_store(v, &dist[((size_t)b * NN + gi) * NN + gj]);
        }
    }
}

// ---------------- top-k (k=20) per row ----------------
__global__ __launch_bounds__(256)
void topk_kernel(const float* __restrict__ dist, int* __restrict__ idx) {
    int row = blockIdx.x * blockDim.x + threadIdx.x;
    if (row >= BB * NN) return;
    const float* d = dist + (size_t)row * NN;
    float v[KNN]; int ix[KNN];
#pragma unroll
    for (int t = 0; t < KNN; ++t) { v[t] = -3.4e38f; ix[t] = 0; }
    for (int j = 0; j < NN; ++j) {
        float x = __builtin_nontemporal_load(d + j);
        if (x > v[KNN - 1]) {
            v[KNN - 1] = x; ix[KNN - 1] = j;
#pragma unroll
            for (int t = KNN - 1; t > 0; --t) {
                if (v[t] > v[t - 1]) {
                    float tv = v[t]; v[t] = v[t - 1]; v[t - 1] = tv;
                    int ti = ix[t]; ix[t] = ix[t - 1]; ix[t - 1] = ti;
                }
            }
        }
    }
#pragma unroll
    for (int t = 0; t < KNN; ++t) idx[(size_t)row * KNN + t] = ix[t];
}

// ---------------- EdgeConv:  y = Wd . nbr + (Wc-Wd) . ctr  (double-buffered) ----------------
// mode 0: per-channel sum/sumsq of pre-BN y.  mode 1: BN+ReLU+max over k -> f16 out slice.
__global__ __launch_bounds__(256)
void edgeconv_kernel(const _Float16* __restrict__ feat, int stride, int chanoff, int Cpad,
                     const int* __restrict__ idx,
                     const _Float16* __restrict__ wd, const _Float16* __restrict__ wdel,
                     float* __restrict__ sum, float* __restrict__ sumsq,
                     const float* __restrict__ scale, const float* __restrict__ shift,
                     _Float16* __restrict__ outp, int ochanoff, int mode) {
    __shared__ __align__(64) _Float16 fs[2 * 320 * 32];   // gathered neighbor rows
    __shared__ __align__(64) _Float16 cs[2 * 16 * 32];    // center rows
    __shared__ __align__(64) _Float16 wsd[2 * 32 * 32];   // Wd chunk
    __shared__ __align__(64) _Float16 wse[2 * 32 * 32];   // (Wc-Wd) chunk
    __shared__ float ys[16 * 32];                         // center-term result
    __shared__ unsigned int outs[16 * 32];                // uint-punned max (post-ReLU >= 0)
    __shared__ int idx_s[320];
    int b = blockIdx.z;
    int pbase = blockIdx.x * 16;
    int obase = blockIdx.y * 32;
    int tid = threadIdx.x, lane = tid & 31, wave = tid >> 5;

    const int* irow = idx + ((size_t)b * NN + pbase) * KNN;
    for (int f = tid; f < 320; f += 256) idx_s[f] = irow[f];
    if (mode == 1)
        for (int f = tid; f < 16 * 32; f += 256) outs[f] = 0u;
    __syncthreads();                                      // idx_s visible before first issue

    v8f acc[5];
#pragma unroll
    for (int s = 0; s < 5; ++s) acc[s] = (v8f){};
    v8f accc = {};                                        // center accumulator (waves 0,1)

    const _Float16* fb = feat + (size_t)b * NN * stride + chanoff;
    int nchunks = Cpad >> 5;

    auto issue = [&](int kc, int p) {                     // 6 async copies / thread / chunk
        int c0 = kc << 5;
        _Float16* fsp = fs + p * 10240;
#pragma unroll
        for (int s = 0; s < 5; ++s) {
            int f = tid + 256 * s;                        // 320 edges x 4 quads
            int e = f >> 2, q = (f & 3) * 8;
            int j = idx_s[e];
            async_cp16(&fsp[e * 32 + q], fb + (size_t)j * stride + c0 + q);
        }
        if (tid < 64) {                                   // waves 0-1: center rows
            int pp = tid >> 2, q = (tid & 3) * 8;
            async_cp16(&cs[p * 512 + pp * 32 + q], fb + (size_t)(pbase + pp) * stride + c0 + q);
        } else if (tid < 192) {                           // waves 2-5: Wd tile
            int f = tid - 64;
            int o = f >> 2, q = (f & 3) * 8;
            async_cp16(&wsd[p * 1024 + o * 32 + q], wd + (size_t)(obase + o) * Cpad + c0 + q);
        } else {                                          // waves 6-7: Wdelta tile
            int f = tid - 192;
            int o = f >> 2, q = (f & 3) * 8;
            async_cp16(&wse[p * 1024 + o * 32 + q], wdel + (size_t)(obase + o) * Cpad + c0 + q);
        }
    };
    issue(0, 0);
    for (int kc = 0; kc < nchunks; ++kc) {
        int p = kc & 1;
        __syncthreads();
        if (kc + 1 < nchunks) { issue(kc + 1, p ^ 1); async_wait6(); }
        else                  { async_wait0(); }
        __syncthreads();
        const _Float16* fsp = fs + p * 10240;
#pragma unroll
        for (int s = 0; s < 5; ++s) {                     // 40 tiles (20 M x 2 N) / 8 waves
            int t = wave + 8 * s;
            int nt = t / 20, mt = t - nt * 20;
            acc[s] = wmma16(load_a_frag(fsp + mt * 16 * 32, 32, lane),
                            load_b_frag(wsd + p * 1024 + nt * 16 * 32, 32, lane), acc[s]);
        }
        if (wave < 2)                                     // center term, one tile per ntile
            accc = wmma16(load_a_frag(cs + p * 512, 32, lane),
                          load_b_frag(wse + p * 1024 + wave * 16 * 32, 32, lane), accc);
    }
    __syncthreads();
    int hi = (lane >> 4) * 8, col = lane & 15;
    if (wave < 2) {
#pragma unroll
        for (int rr = 0; rr < 8; ++rr)
            ys[(rr + hi) * 32 + wave * 16 + col] = accc[rr];
    }
    __syncthreads();

    if (mode == 0) {
#pragma unroll
        for (int s = 0; s < 5; ++s) {
            int t = wave + 8 * s;
            int nt = t / 20, mt = t - nt * 20;
            int ol = nt * 16 + col;
            float s1 = 0.f, s2 = 0.f;
#pragma unroll
            for (int rr = 0; rr < 8; ++rr) {
                int m = mt * 16 + rr + hi;
                int pp = m / KNN;
                float y = acc[s][rr] + ys[pp * 32 + ol];
                s1 += y; s2 += y * y;
            }
            s1 += __shfl_xor(s1, 16);
            s2 += __shfl_xor(s2, 16);
            if (lane < 16) {
                int o = obase + nt * 16 + lane;
                atomicAdd(&sum[o], s1);
                atomicAdd(&sumsq[o], s2);
            }
        }
    } else {
#pragma unroll
        for (int s = 0; s < 5; ++s) {
            int t = wave + 8 * s;
            int nt = t / 20, mt = t - nt * 20;
            int ol = nt * 16 + col;
            int o = obase + ol;
            float sc = scale[o], sh = shift[o];
#pragma unroll
            for (int rr = 0; rr < 8; ++rr) {
                int m = mt * 16 + rr + hi;
                int pp = m / KNN;
                float y = (acc[s][rr] + ys[pp * 32 + ol]) * sc + sh;
                y = y > 0.f ? y : 0.f;
                atomicMax(&outs[pp * 32 + ol], __float_as_uint(y));
            }
        }
        __syncthreads();
        for (int f = tid; f < 16 * 32; f += 256) {
            int pp = f >> 5, c = f & 31;
            outp[((size_t)b * NN + pbase + pp) * 512 + ochanoff + obase + c] =
                (_Float16)__uint_as_float(outs[f]);
        }
    }
}

// ---------------- final: y = W5 . xc row, BN + ReLU + max over n (double-buffered) ----------------
__global__ __launch_bounds__(256)
void final_kernel(const _Float16* __restrict__ xch, const _Float16* __restrict__ w5h,
                  float* __restrict__ sum, float* __restrict__ sumsq,
                  const float* __restrict__ scale, const float* __restrict__ shift,
                  float* __restrict__ out, int mode) {
    __shared__ __align__(64) _Float16 xs[2 * 64 * 32];
    __shared__ __align__(64) _Float16 wsh[2 * 64 * 32];
    int b = blockIdx.z;
    int nbase = blockIdx.x * 64;
    int obase = blockIdx.y * 64;
    int tid = threadIdx.x, lane = tid & 31, wave = tid >> 5;
    v8f acc0 = {}, acc1 = {};
    const int t0 = wave, t1 = wave + 8;
    int r = tid >> 2, q = (tid & 3) * 8;

    auto issue = [&](int kc, int p) {                     // 2 async copies / thread / chunk
        int c0 = kc << 5;
        async_cp16(&xs[p * 2048 + r * 32 + q], xch + ((size_t)b * NN + nbase + r) * 512 + c0 + q);
        async_cp16(&wsh[p * 2048 + r * 32 + q], w5h + (size_t)(obase + r) * 512 + c0 + q);
    };
    issue(0, 0);
    for (int kc = 0; kc < 16; ++kc) {                     // 512 / 32
        int p = kc & 1;
        __syncthreads();
        if (kc + 1 < 16) { issue(kc + 1, p ^ 1); async_wait2(); }
        else             { async_wait0(); }
        __syncthreads();
        const _Float16* bx = xs + p * 2048;
        const _Float16* bw = wsh + p * 2048;
        {
            int ti = t0 >> 2, tj = t0 & 3;
            acc0 = wmma16(load_a_frag(bx + ti * 16 * 32, 32, lane),
                          load_b_frag(bw + tj * 16 * 32, 32, lane), acc0);
        }
        {
            int ti = t1 >> 2, tj = t1 & 3;
            acc1 = wmma16(load_a_frag(bx + ti * 16 * 32, 32, lane),
                          load_b_frag(bw + tj * 16 * 32, 32, lane), acc1);
        }
    }
    int col = lane & 15;
#pragma unroll
    for (int s = 0; s < 2; ++s) {
        v8f acc = s ? acc1 : acc0;
        int t = s ? t1 : t0;
        int tj = t & 3;
        int o = obase + tj * 16 + col;
        if (mode == 0) {
            float s1 = 0.f, s2 = 0.f;
#pragma unroll
            for (int rr = 0; rr < 8; ++rr) { float y = acc[rr]; s1 += y; s2 += y * y; }
            s1 += __shfl_xor(s1, 16);
            s2 += __shfl_xor(s2, 16);
            if (lane < 16) { atomicAdd(&sum[o], s1); atomicAdd(&sumsq[o], s2); }
        } else {
            float sc = scale[o], sh = shift[o];
            float m = 0.f;
#pragma unroll
            for (int rr = 0; rr < 8; ++rr) {
                float y = acc[rr] * sc + sh;
                m = fmaxf(m, y > 0.f ? y : 0.f);
            }
            m = fmaxf(m, __shfl_xor(m, 16));
            if (lane < 16)
                atomicMax((unsigned int*)&out[(size_t)b * 1024 + o], __float_as_uint(m));
        }
    }
}

// ---------------- host orchestration ----------------
extern "C" void kernel_launch(void* const* d_in, const int* in_sizes, int n_in,
                              void* d_out, int out_size, void* d_ws, size_t ws_size,
                              hipStream_t stream) {
    (void)in_sizes; (void)n_in; (void)out_size; (void)ws_size;
    const float* x  = (const float*)d_in[0];
    const float* Wl[5] = { (const float*)d_in[1], (const float*)d_in[2], (const float*)d_in[3],
                           (const float*)d_in[4], (const float*)d_in[5] };
    const float* gl[5] = { (const float*)d_in[6],  (const float*)d_in[8],  (const float*)d_in[10],
                           (const float*)d_in[12], (const float*)d_in[14] };
    const float* bl[5] = { (const float*)d_in[7],  (const float*)d_in[9],  (const float*)d_in[11],
                           (const float*)d_in[13], (const float*)d_in[15] };

    // workspace layout (fp32 region, then int, then f16 region; all 16B aligned)
    float* dist  = (float*)d_ws;                         // B*N*N
    float* sq    = dist + (size_t)BB * NN * NN;          // B*N
    float* stats = sq + (size_t)BB * NN;                 // 4096
    float* sum = stats, *sumsq = stats + 1024, *scale = stats + 2048, *shift = stats + 3072;
    int* idxb = (int*)(stats + 4096);                    // B*N*KNN
    _Float16* xch = (_Float16*)(idxb + (size_t)BB * NN * KNN);  // B*N*512
    _Float16* xh0 = xch + (size_t)BB * NN * 512;         // B*N*32
    _Float16* w5h = xh0 + (size_t)BB * NN * 32;          // 1024*512
    _Float16* wdh = w5h + (size_t)1024 * 512;            // <= 256*128
    _Float16* wdeh = wdh + (size_t)256 * 128;

    const int threads = 256;
    const int rowBlocks = (BB * NN) / threads;           // 128

    conv_in_kernel<<<rowBlocks, threads, 0, stream>>>(x, xh0);

    const _Float16* inp[4] = { xh0, xch, xch, xch };
    const int instr[4]  = { 32, 512, 512, 512 };
    const int inoff[4]  = { 0, 0, 64, 128 };
    const int Cin[4]    = { 3, 64, 64, 128 };
    const int Cpad[4]   = { 32, 64, 64, 128 };
    const int Oc[4]     = { 64, 64, 128, 256 };
    const int ooff[4]   = { 0, 64, 128, 256 };
    const float invcnt_e = 1.0f / (float)((size_t)BB * NN * KNN);

    for (int l = 0; l < 4; ++l) {
        prep_w_kernel<<<(Oc[l] * Cpad[l]) / threads, threads, 0, stream>>>(
            Wl[l], wdh, wdeh, Oc[l], Cin[l], Cpad[l]);
        sqnorm_kernel<<<rowBlocks, threads, 0, stream>>>(inp[l], instr[l], inoff[l], Cpad[l], sq);
        dist_kernel<<<dim3(NN / 64, NN / 64, BB), threads, 0, stream>>>(
            inp[l], instr[l], inoff[l], Cpad[l], sq, dist);
        topk_kernel<<<rowBlocks, threads, 0, stream>>>(dist, idxb);
        fill_kernel<<<8, threads, 0, stream>>>(sum, 2048, 0.f);
        edgeconv_kernel<<<dim3(NN / 16, Oc[l] / 32, BB), threads, 0, stream>>>(
            inp[l], instr[l], inoff[l], Cpad[l], idxb, wdh, wdeh,
            sum, sumsq, scale, shift, xch, ooff[l], /*mode=*/0);
        bn_finalize_kernel<<<(Oc[l] + 63) / 64, 64, 0, stream>>>(
            sum, sumsq, gl[l], bl[l], scale, shift, Oc[l], invcnt_e);
        edgeconv_kernel<<<dim3(NN / 16, Oc[l] / 32, BB), threads, 0, stream>>>(
            inp[l], instr[l], inoff[l], Cpad[l], idxb, wdh, wdeh,
            sum, sumsq, scale, shift, xch, ooff[l], /*mode=*/1);
    }

    // final layer: (B*N, 512) x (512, 1024), BN over (b,n), ReLU, max over n
    const float invcnt_f = 1.0f / (float)((size_t)BB * NN);
    conv_w5_kernel<<<(1024 * 512) / threads, threads, 0, stream>>>(Wl[4], w5h);
    fill_kernel<<<8, threads, 0, stream>>>(sum, 2048, 0.f);
    final_kernel<<<dim3(NN / 64, 1024 / 64, BB), threads, 0, stream>>>(
        xch, w5h, sum, sumsq, scale, shift, (float*)d_out, /*mode=*/0);
    bn_finalize_kernel<<<16, 64, 0, stream>>>(sum, sumsq, gl[4], bl[4], scale, shift, 1024, invcnt_f);
    fill_kernel<<<(BB * 1024) / threads, threads, 0, stream>>>((float*)d_out, BB * 1024, 0.f);
    final_kernel<<<dim3(NN / 64, 1024 / 64, BB), threads, 0, stream>>>(
        xch, w5h, sum, sumsq, scale, shift, (float*)d_out, /*mode=*/1);
}
